// SPPGN1_72610717106394
// MI455X (gfx1250) — matrix operand order repaired and checked by reading, not compiled, not run on hardware
//
#include <hip/hip_runtime.h>
#include <hip/hip_bf16.h>
#include <math.h>

// ---------------- sizes (match reference) ----------------
#define PP 262144
#define TT 1048576
#define NN 65536
#define GG 512
#define HH 128
#define LL 3

typedef _Float16 h16;
typedef __attribute__((ext_vector_type(16))) _Float16 v16h;
typedef __attribute__((ext_vector_type(8)))  _Float16 v8h;
typedef __attribute__((ext_vector_type(4)))  _Float16 v4h;
typedef __attribute__((ext_vector_type(8)))  float    v8f;

static __device__ __forceinline__ float fatomic_add(float* p, float v) {
  return __hip_atomic_fetch_add(p, v, __ATOMIC_RELAXED, __HIP_MEMORY_SCOPE_AGENT);
}

// ---------------------------------------------------------------------------
// Fused MLP: Y = relu( (X @ Wa) * scale + shift ) @ Wb + bb (+ resid)
// X is [P, KA] fp32 (KA = 128 or 256; cols >=128 come from X1), W*T are f16
// transposed weights (N-major: WT[n*KA + k]). One wave = 16 output rows.
// 8 waves / block = 128 rows / block. KA is a template param so both GEMM
// k-loops fully unroll (static wmma chain, max memory-level parallelism).
// ---------------------------------------------------------------------------
template <int KA>
__launch_bounds__(256, 2)
__global__ void mlp_fused(const float* __restrict__ X0, const float* __restrict__ X1,
                          const h16* __restrict__ WaT,
                          const float* __restrict__ scale, const float* __restrict__ shift,
                          const h16* __restrict__ WbT,
                          const float* __restrict__ bb,
                          const float* __restrict__ resid,
                          float* __restrict__ outF, h16* __restrict__ outH)
{
  __shared__ h16 hbuf[8 * 16 * 128];   // 32 KB: per-wave 16x128 f16 hidden tile

  const int lane = threadIdx.x & 31;
  const int wave = threadIdx.x >> 5;
  const int m    = lane & 15;          // A-row / B-col within tile
  const int hi   = lane >> 4;          // half-wave selector
  const int row0 = blockIdx.x * 128 + wave * 16;
  const int arow = row0 + m;

  // ---------------- GEMM1: hidden = X @ Wa ----------------
  v8f acc[8] = {};
  #pragma unroll
  for (int kc = 0; kc < KA; kc += 32) {
    // A tile, 16-bit A layout: lane<16 -> K = kc + {0..7, 16..23}; lane>=16 -> +8
    const int c0 = kc + (hi ? 8 : 0);
    const int c1 = c0 + 16;
    v16h a;
    {
      // 8-elem chunks never straddle the 128 boundary (kc mult of 32, offsets mult of 8)
      const float* s0 = (kc < 128) ? (X0 + (size_t)arow * 128 + c0)
                                   : (X1 + (size_t)arow * 128 + (c0 - 128));
      const float* s1 = (kc < 128) ? (X0 + (size_t)arow * 128 + c1)
                                   : (X1 + (size_t)arow * 128 + (c1 - 128));
      const float4 f0 = ((const float4*)s0)[0];
      const float4 f1 = ((const float4*)s0)[1];
      const float4 f2 = ((const float4*)s1)[0];
      const float4 f3 = ((const float4*)s1)[1];
      a[0]=(h16)f0.x; a[1]=(h16)f0.y; a[2]=(h16)f0.z; a[3]=(h16)f0.w;
      a[4]=(h16)f1.x; a[5]=(h16)f1.y; a[6]=(h16)f1.z; a[7]=(h16)f1.w;
      a[8]=(h16)f2.x; a[9]=(h16)f2.y; a[10]=(h16)f2.z; a[11]=(h16)f2.w;
      a[12]=(h16)f3.x; a[13]=(h16)f3.y; a[14]=(h16)f3.z; a[15]=(h16)f3.w;
    }
    #pragma unroll
    for (int nt = 0; nt < 8; ++nt) {
      // B tile, 16-bit B layout: lane holds col n, K = kc + {0..7, 16..23}; lane>=16 -> +8
      const h16* bp = WaT + (size_t)(nt * 16 + m) * KA + kc + (hi ? 8 : 0);
      const v8h b0 = *(const v8h*)(bp);
      const v8h b1 = *(const v8h*)(bp + 16);
      v16h b;
      #pragma unroll
      for (int i = 0; i < 8; ++i) { b[i] = b0[i]; b[8 + i] = b1[i]; }
      acc[nt] = __builtin_amdgcn_wmma_f32_16x16x32_f16(
          false, a, false, b, (short)0, acc[nt], false, false);
    }
  }

  // ---------------- BN affine + ReLU -> LDS (f16, row-major) ----------------
  h16* hw = hbuf + wave * (16 * 128);
  #pragma unroll
  for (int nt = 0; nt < 8; ++nt) {
    const int n = nt * 16 + m;
    const float sc = scale[n];
    const float sh = shift[n];
    #pragma unroll
    for (int r = 0; r < 8; ++r) {
      const int mm = r + hi * 8;     // C-layout row
      float v = acc[nt][r] * sc + sh;
      v = v > 0.0f ? v : 0.0f;
      hw[mm * 128 + n] = (h16)v;
    }
  }
  __syncthreads();

  // ---------------- GEMM2: Y = hidden @ Wb ----------------
  v8f acc2[8] = {};
  #pragma unroll
  for (int kc = 0; kc < 128; kc += 32) {
    const int c0 = kc + (hi ? 8 : 0);
    v16h a;
    {
      const v8h a0 = *(const v8h*)(hw + m * 128 + c0);
      const v8h a1 = *(const v8h*)(hw + m * 128 + c0 + 16);
      #pragma unroll
      for (int i = 0; i < 8; ++i) { a[i] = a0[i]; a[8 + i] = a1[i]; }
    }
    #pragma unroll
    for (int nt = 0; nt < 8; ++nt) {
      const h16* bp = WbT + (size_t)(nt * 16 + m) * 128 + kc + (hi ? 8 : 0);
      const v8h b0 = *(const v8h*)(bp);
      const v8h b1 = *(const v8h*)(bp + 16);
      v16h b;
      #pragma unroll
      for (int i = 0; i < 8; ++i) { b[i] = b0[i]; b[8 + i] = b1[i]; }
      acc2[nt] = __builtin_amdgcn_wmma_f32_16x16x32_f16(
          false, a, false, b, (short)0, acc2[nt], false, false);
    }
  }

  // ---------------- epilogue: bias (+resid), store f32 and/or f16 ----------
  #pragma unroll
  for (int nt = 0; nt < 8; ++nt) {
    const int n = nt * 16 + m;
    const float bias = bb[n];
    #pragma unroll
    for (int r = 0; r < 8; ++r) {
      const int grow = row0 + r + hi * 8;
      float y = acc2[nt][r] + bias;
      if (resid) y += resid[(size_t)grow * 128 + n];
      if (outF)  outF[(size_t)grow * 128 + n] = y;
      if (outH)  outH[(size_t)grow * 128 + n] = (h16)y;
    }
  }
}

// ---------------------------------------------------------------------------
// x3_agg[idx0] += x2_1[idx1] * x2_2[idx2]   (one wave per tuple, 4 cols/lane)
// ---------------------------------------------------------------------------
__launch_bounds__(256)
__global__ void gms_kernel(const h16* __restrict__ xa, const h16* __restrict__ xb,
                           const int* __restrict__ tindex, float* __restrict__ out)
{
  const int t = blockIdx.x * 8 + (threadIdx.x >> 5);
  const int c = (threadIdx.x & 31) * 4;
  const int i0 = tindex[t];
  const int i1 = tindex[TT + t];
  const int i2 = tindex[2 * TT + t];
  const v4h a = *(const v4h*)(xa + (size_t)i1 * 128 + c);
  const v4h b = *(const v4h*)(xb + (size_t)i2 * 128 + c);
  float* o = out + (size_t)i0 * 128 + c;
  #pragma unroll
  for (int j = 0; j < 4; ++j)
    fatomic_add(o + j, (float)a[j] * (float)b[j]);
}

__launch_bounds__(256)
__global__ void ssqrt_kernel(float* __restrict__ x, int n)
{
  const int i = (blockIdx.x * 256 + threadIdx.x) * 4;
  if (i >= n) return;
  #pragma unroll
  for (int j = 0; j < 4; ++j) {
    const float v = x[i + j];
    x[i + j] = v > 0.0f ? sqrtf(v) : -sqrtf(-v);
  }
}

__launch_bounds__(256)
__global__ void zero_kernel(float* __restrict__ p, int n)
{
  const int i = (blockIdx.x * 256 + threadIdx.x) * 4;
  if (i >= n) return;
  #pragma unroll
  for (int j = 0; j < 4; ++j) p[i + j] = 0.0f;
}

// agg2[pair_batch[p]] += x2[p]
__launch_bounds__(256)
__global__ void pool_pair(const float* __restrict__ x2, const int* __restrict__ pb,
                          float* __restrict__ agg2)
{
  const int p = blockIdx.x * 8 + (threadIdx.x >> 5);
  const int c = (threadIdx.x & 31) * 4;
  const int g = pb[p];
  const float4 v = *(const float4*)(x2 + (size_t)p * 128 + c);
  float* o = agg2 + (size_t)g * 128 + c;
  fatomic_add(o + 0, v.x); fatomic_add(o + 1, v.y);
  fatomic_add(o + 2, v.z); fatomic_add(o + 3, v.w);
}

// agg1[node_batch[i]] += x2[loop_idx[i]]
__launch_bounds__(256)
__global__ void pool_node(const float* __restrict__ x2, const int* __restrict__ nb,
                          const int* __restrict__ li, float* __restrict__ agg1)
{
  const int i = blockIdx.x * 8 + (threadIdx.x >> 5);
  const int c = (threadIdx.x & 31) * 4;
  const int g = nb[i];
  const int s = li[i];
  const float4 v = *(const float4*)(x2 + (size_t)s * 128 + c);
  float* o = agg1 + (size_t)g * 128 + c;
  fatomic_add(o + 0, v.x); fatomic_add(o + 1, v.y);
  fatomic_add(o + 2, v.z); fatomic_add(o + 3, v.w);
}

// out[g] = sum_h agg1[g][h]*W[h] + (agg2[g][h]-agg1[g][h])*W[128+h] + b[0]
__launch_bounds__(256)
__global__ void final_kernel(const float* __restrict__ agg1, const float* __restrict__ agg2,
                             const float* __restrict__ W, const float* __restrict__ b,
                             float* __restrict__ out)
{
  const int g = blockIdx.x * 8 + (threadIdx.x >> 5);
  const int lane = threadIdx.x & 31;
  float s = 0.0f;
  for (int h = lane; h < 128; h += 32) {
    const float a1 = agg1[(size_t)g * 128 + h];
    const float a2 = agg2[(size_t)g * 128 + h];
    s += a1 * W[h] + (a2 - a1) * W[128 + h];
  }
  #pragma unroll
  for (int off = 16; off > 0; off >>= 1) s += __shfl_xor(s, off, 32);
  if (lane == 0) out[g] = s + b[0];
}

// WT[n*K + k] = (f16) W[k*N + n]
__launch_bounds__(256)
__global__ void transpose_h16(const float* __restrict__ W, h16* __restrict__ WT,
                              int K, int N)
{
  const int i = blockIdx.x * 256 + threadIdx.x;
  if (i >= K * N) return;
  const int k = i / N, n = i % N;
  WT[(size_t)n * K + k] = (h16)W[(size_t)k * N + n];
}

// scale = g * rsqrt(v + eps); shift = be + (ba - m) * scale
__launch_bounds__(128)
__global__ void bnfold_kernel(const float* __restrict__ g, const float* __restrict__ be,
                              const float* __restrict__ m, const float* __restrict__ v,
                              const float* __restrict__ ba,
                              float* __restrict__ sc, float* __restrict__ sh)
{
  const int i = threadIdx.x;
  const float s = g[i] * rsqrtf(v[i] + 1e-5f);
  sc[i] = s;
  sh[i] = be[i] + (ba[i] - m[i]) * s;
}

// ---------------------------------------------------------------------------
extern "C" void kernel_launch(void* const* d_in, const int* in_sizes, int n_in,
                              void* d_out, int out_size, void* d_ws, size_t ws_size,
                              hipStream_t stream)
{
  (void)in_sizes; (void)n_in; (void)out_size; (void)ws_size;

  const float* pair_h     = (const float*)d_in[0];
  const int*   tindex     = (const int*)  d_in[1];
  const int*   pair_batch = (const int*)  d_in[2];
  const int*   node_batch = (const int*)  d_in[3];
  const int*   loop_idx   = (const int*)  d_in[4];
  const float* W1a = (const float*)d_in[5];
  const float* b1a = (const float*)d_in[6];
  const float* g1  = (const float*)d_in[7];
  const float* be1 = (const float*)d_in[8];
  const float* m1  = (const float*)d_in[9];
  const float* v1  = (const float*)d_in[10];
  const float* W1b = (const float*)d_in[11];
  const float* b1b = (const float*)d_in[12];
  const float* W2a = (const float*)d_in[13];
  const float* b2a = (const float*)d_in[14];
  const float* g2  = (const float*)d_in[15];
  const float* be2 = (const float*)d_in[16];
  const float* m2  = (const float*)d_in[17];
  const float* v2  = (const float*)d_in[18];
  const float* W2b = (const float*)d_in[19];
  const float* b2b = (const float*)d_in[20];
  const float* Wu1 = (const float*)d_in[21];
  const float* bu1 = (const float*)d_in[22];
  const float* gu  = (const float*)d_in[23];
  const float* beu = (const float*)d_in[24];
  const float* mu  = (const float*)d_in[25];
  const float* vu  = (const float*)d_in[26];
  const float* Wu2 = (const float*)d_in[27];
  const float* bu2 = (const float*)d_in[28];
  const float* Wout = (const float*)d_in[29];
  const float* bout = (const float*)d_in[30];

  // ---- workspace carve ----
  char* w = (char*)d_ws;
  auto carve = [&](size_t bytes) -> void* {
    void* p = (void*)w;
    w += (bytes + 255) & ~(size_t)255;
    return p;
  };
  float* x2   = (float*)carve((size_t)PP * HH * 4);
  float* x3   = (float*)carve((size_t)PP * HH * 4);
  h16*   x21  = (h16*)  carve((size_t)PP * HH * 2);
  h16*   x22  = (h16*)  carve((size_t)PP * HH * 2);
  h16* W1aT = (h16*)carve((size_t)LL * HH * HH * 2);
  h16* W1bT = (h16*)carve((size_t)LL * HH * HH * 2);
  h16* W2aT = (h16*)carve((size_t)LL * HH * HH * 2);
  h16* W2bT = (h16*)carve((size_t)LL * HH * HH * 2);
  h16* Wu2T = (h16*)carve((size_t)LL * HH * HH * 2);
  h16* Wu1T = (h16*)carve((size_t)LL * 2 * HH * HH * 2);
  float* sc1 = (float*)carve((size_t)LL * HH * 4);
  float* sh1 = (float*)carve((size_t)LL * HH * 4);
  float* sc2 = (float*)carve((size_t)LL * HH * 4);
  float* sh2 = (float*)carve((size_t)LL * HH * 4);
  float* scu = (float*)carve((size_t)LL * HH * 4);
  float* shu = (float*)carve((size_t)LL * HH * 4);
  float* agg1 = (float*)carve((size_t)GG * HH * 4);
  float* agg2 = (float*)carve((size_t)GG * HH * 4);

  // ---- weight prep: f16 transposes + BN folds ----
  for (int l = 0; l < LL; ++l) {
    const size_t o = (size_t)l * HH * HH;
    transpose_h16<<<64, 256, 0, stream>>>(W1a + o, W1aT + o, HH, HH);
    transpose_h16<<<64, 256, 0, stream>>>(W1b + o, W1bT + o, HH, HH);
    transpose_h16<<<64, 256, 0, stream>>>(W2a + o, W2aT + o, HH, HH);
    transpose_h16<<<64, 256, 0, stream>>>(W2b + o, W2bT + o, HH, HH);
    transpose_h16<<<64, 256, 0, stream>>>(Wu2 + o, Wu2T + o, HH, HH);
    transpose_h16<<<128, 256, 0, stream>>>(Wu1 + 2 * o, Wu1T + 2 * o, 2 * HH, HH);
    const size_t oh = (size_t)l * HH;
    bnfold_kernel<<<1, 128, 0, stream>>>(g1 + oh, be1 + oh, m1 + oh, v1 + oh, b1a + oh, sc1 + oh, sh1 + oh);
    bnfold_kernel<<<1, 128, 0, stream>>>(g2 + oh, be2 + oh, m2 + oh, v2 + oh, b2a + oh, sc2 + oh, sh2 + oh);
    bnfold_kernel<<<1, 128, 0, stream>>>(gu + oh, beu + oh, mu + oh, vu + oh, bu1 + oh, scu + oh, shu + oh);
  }

  // ---- 3 PPGN layers ----
  for (int l = 0; l < LL; ++l) {
    const size_t o  = (size_t)l * HH * HH;
    const size_t oh = (size_t)l * HH;
    const float* xin = (l == 0) ? pair_h : x2;

    // mlp1 / mlp2 -> f16 outputs for the gather stage
    mlp_fused<128><<<PP / 128, 256, 0, stream>>>(xin, nullptr, W1aT + o,
        sc1 + oh, sh1 + oh, W1bT + o, b1b + oh, nullptr, nullptr, x21);
    mlp_fused<128><<<PP / 128, 256, 0, stream>>>(xin, nullptr, W2aT + o,
        sc2 + oh, sh2 + oh, W2bT + o, b2b + oh, nullptr, nullptr, x22);

    // x3_agg = segment_sum(x2_1[idx1] * x2_2[idx2], idx0); then signed sqrt
    zero_kernel<<<(PP * HH) / 1024, 256, 0, stream>>>(x3, PP * HH);
    gms_kernel<<<TT / 8, 256, 0, stream>>>(x21, x22, tindex, x3);
    ssqrt_kernel<<<(PP * HH) / 1024, 256, 0, stream>>>(x3, PP * HH);

    // update MLP on [xin | x3], residual add, f32 output -> x2 (in-place safe)
    mlp_fused<256><<<PP / 128, 256, 0, stream>>>(xin, x3, Wu1T + 2 * o,
        scu + oh, shu + oh, Wu2T + o, bu2 + oh, xin, x2, nullptr);
  }

  // ---- pooling + output head ----
  zero_kernel<<<(GG * HH) / 1024, 256, 0, stream>>>(agg1, GG * HH);
  zero_kernel<<<(GG * HH) / 1024, 256, 0, stream>>>(agg2, GG * HH);
  pool_pair<<<PP / 8, 256, 0, stream>>>(x2, pair_batch, agg2);
  pool_node<<<NN / 8, 256, 0, stream>>>(x2, node_batch, loop_idx, agg1);
  final_kernel<<<GG / 8, 256, 0, stream>>>(agg1, agg2, Wout, bout, (float*)d_out);
}
